// LocalCausalSelfAttention_46677704573179
// MI455X (gfx1250) — matrix operand order repaired and checked by reading, not compiled, not run on hardware
//
#include <hip/hip_runtime.h>

// ---------------------------------------------------------------------------
// BigBird full self-attention for MI455X (gfx1250).
// bf16 WMMA pipeline + async global->LDS staging (ASYNCcnt) + ones-WMMA rowsum.
// ---------------------------------------------------------------------------

typedef __attribute__((ext_vector_type(16))) __bf16 bf16x16;
typedef __attribute__((ext_vector_type(8)))  __bf16 bf16x8;
typedef __attribute__((ext_vector_type(8)))  float  f32x8;

#define B_   4
#define L_   2048
#define D_   1024
#define H_   16
#define HD_  64
#define MTOK (B_ * L_)   // 8192 tokens

// --------------------------- WMMA helpers ----------------------------------

__device__ __forceinline__ f32x8 wmma_bf16(bf16x16 a, bf16x16 b, f32x8 c) {
  return __builtin_amdgcn_wmma_f32_16x16x32_bf16(
      /*neg_a=*/false, a, /*neg_b=*/false, b,
      /*c_mod=*/(short)0, c, /*reuse_a=*/false, /*reuse_b=*/false);
}

// A-fragment (16x32, 16-bit): lane m (0..15): K {k0..k0+7, k0+16..k0+23};
// lane m+16: K {k0+8..k0+15, k0+24..k0+31}.  prow = this lane's row base.
__device__ __forceinline__ bf16x16 frag_a(const __bf16* prow, int k0, int lane) {
  const int sel = (lane >> 4) & 1;
  const __bf16* p = prow + k0 + sel * 8;
  union { bf16x16 v; bf16x8 h[2]; } u;
  u.h[0] = *(const bf16x8*)(p);
  u.h[1] = *(const bf16x8*)(p + 16);
  return u.v;
}

// B-fragment (32x16, 16-bit): lane n: K {k0..k0+7, k0+16..k0+23}; lane n+16: K {k0+8..k0+15, k0+24..k0+31}.
__device__ __forceinline__ bf16x16 frag_b(const __bf16* prow, int k0, int lane) {
  const int sel = (lane >> 4) & 1;
  const __bf16* p = prow + k0 + sel * 8;
  union { bf16x16 v; bf16x8 h[2]; } u;
  u.h[0] = *(const bf16x8*)(p);
  u.h[1] = *(const bf16x8*)(p + 16);
  return u.v;
}

// Async DMA: one 16-byte chunk global -> LDS, tracked by ASYNCcnt.
__device__ __forceinline__ void async_b128(unsigned lds_byte_off, const void* gaddr) {
  asm volatile("global_load_async_to_lds_b128 %0, %1, off"
               :: "v"(lds_byte_off), "v"(gaddr) : "memory");
}
__device__ __forceinline__ void wait_async0() {
  asm volatile("s_wait_asynccnt 0" ::: "memory");
}
__device__ __forceinline__ unsigned lds_off(const void* p) {
  return (unsigned)(size_t)p;   // LDS flat addresses carry the offset in [31:0]
}

// --------------------------- f32 -> bf16 convert ---------------------------

__global__ __launch_bounds__(256) void cvt_f32_bf16_kernel(
    const float* __restrict__ in, __bf16* __restrict__ out, int n) {
  int i = (blockIdx.x * 256 + threadIdx.x) * 4;
  if (i + 3 < n) {
    float4 f = *(const float4*)(in + i);
    typedef __attribute__((ext_vector_type(4))) __bf16 bf16x4;
    bf16x4 o;
    o[0] = (__bf16)f.x; o[1] = (__bf16)f.y; o[2] = (__bf16)f.z; o[3] = (__bf16)f.w;
    *(bf16x4*)(out + i) = o;
  }
}

// --------------------------- GEMM: C = A * W^T -----------------------------
// A: [MTOK x D_] bf16 row-major.  W: [D_ x D_] bf16 row-major.
// Wave tile 64x64 (16 WMMA per K-step of 32); block = 4 waves (2x2) = 128x128.
// MODE 0: bf16 head-major [b,h,l,hd] * oscale   (Q with 1/8, K with 1)
// MODE 1: bf16 transposed [b,h,hd,l] * oscale   (V)
// MODE 2: f32 row-major [m,n] + bias            (final projection)
template <int MODE>
__global__ __launch_bounds__(128) void gemm_bf16_kernel(
    const __bf16* __restrict__ A, const __bf16* __restrict__ W,
    void* __restrict__ outp, const float* __restrict__ bias, float oscale) {
  const int lane = threadIdx.x & 31;
  const int wid  = threadIdx.x >> 5;
  const int wm   = wid & 1;
  const int wn   = wid >> 1;
  const int m0   = blockIdx.x * 128 + wm * 64;
  const int n0   = blockIdx.y * 128 + wn * 64;
  const int lrow = lane & 15;
  const int half = lane >> 4;

  const __bf16* arow[4];
  const __bf16* brow[4];
#pragma unroll
  for (int t = 0; t < 4; ++t) {
    arow[t] = A + (size_t)(m0 + t * 16 + lrow) * D_;
    brow[t] = W + (size_t)(n0 + t * 16 + lrow) * D_;
  }

  f32x8 acc[4][4] = {};

  for (int k0 = 0; k0 < D_; k0 += 32) {
    if (k0 + 256 < D_) {
      __builtin_prefetch(arow[0] + k0 + 256, 0, 1);   // global_prefetch_b8
      __builtin_prefetch(brow[0] + k0 + 256, 0, 1);
    }
    bf16x16 af[4], bf[4];
#pragma unroll
    for (int t = 0; t < 4; ++t) af[t] = frag_a(arow[t], k0, lane);
#pragma unroll
    for (int t = 0; t < 4; ++t) bf[t] = frag_b(brow[t], k0, lane);
#pragma unroll
    for (int mt = 0; mt < 4; ++mt)
#pragma unroll
      for (int nt = 0; nt < 4; ++nt)
        acc[mt][nt] = wmma_bf16(af[mt], bf[nt], acc[mt][nt]);
  }

  // C layout: element (m_local = r + 8*half, n_local = lane&15) at acc[..][r]
#pragma unroll
  for (int mt = 0; mt < 4; ++mt) {
#pragma unroll
    for (int nt = 0; nt < 4; ++nt) {
      const int n  = n0 + nt * 16 + lrow;
      const int h  = n >> 6;
      const int hd = n & 63;
      float bv = 0.0f;
      if (MODE == 2) bv = bias[n];
#pragma unroll
      for (int r = 0; r < 8; ++r) {
        const int m = m0 + mt * 16 + half * 8 + r;
        const float v = acc[mt][nt][r];
        if (MODE == 0) {
          const int b = m >> 11, l = m & (L_ - 1);
          ((__bf16*)outp)[(((size_t)(b * H_ + h) * L_ + l) << 6) + hd] =
              (__bf16)(v * oscale);
        } else if (MODE == 1) {
          const int b = m >> 11, l = m & (L_ - 1);
          ((__bf16*)outp)[(((size_t)(b * H_ + h) << 6) + hd) * L_ + l] =
              (__bf16)(v * oscale);
        } else {
          ((float*)outp)[(size_t)m * D_ + n] = v + bv;
        }
      }
    }
  }
}

// --------------------------- Flash attention -------------------------------
// Q (pre-scaled by 1/8), K: [b,h,L,HD] bf16.  Vt: [b,h,HD,L] bf16.
// Block = 4 waves, all sharing one (b,h): K/V 64-key tiles are DMA'd to LDS
// with global_load_async_to_lds_b128 (double-buffered, 1 barrier/iter).
// Each wave owns 16 q-rows; online softmax; P row-sums via ones-WMMA;
// P re-laid-out C->A through a private LDS tile.
__global__ __launch_bounds__(128) void attention_kernel(
    const __bf16* __restrict__ Q, const __bf16* __restrict__ K,
    const __bf16* __restrict__ Vt, __bf16* __restrict__ ctx) {
  __shared__ __align__(16) __bf16 ldsK[2][64 * 64];   // [key_local][hd]  8KB x2
  __shared__ __align__(16) __bf16 ldsV[2][64 * 64];   // [hd][key_local]  8KB x2
  __shared__ __align__(16) __bf16 plds[4][16 * 64];   // per-wave P tile  2KB x4

  const int tid   = threadIdx.x;
  const int lane  = tid & 31;
  const int wid   = tid >> 5;
  const int chunk = blockIdx.x * 4 + wid;     // 0 .. 8191
  const int bh    = chunk >> 7;               // 32 blocks per (b,h): no straddle
  const int qb    = (chunk & 127) << 4;
  const int lrow  = lane & 15;
  const int half  = lane >> 4;

  const __bf16* Qb = Q  + (size_t)bh * L_ * HD_;
  const __bf16* Kb = K  + (size_t)bh * L_ * HD_;
  const __bf16* Vb = Vt + (size_t)bh * HD_ * L_;
  __bf16* myp = &plds[wid][0];

  // Issue one 64x64 K tile + 64x64 V tile (16KB) split across 128 threads:
  // 1024 16B chunks, 8 per thread.  chunk c: row = c>>3, sub = c&7.
  auto issue_tile = [&](int ib, int kb) {
#pragma unroll
    for (int i = 0; i < 4; ++i) {
      const int c   = i * 128 + tid;
      const int row = c >> 3, sub = c & 7;
      async_b128(lds_off(&ldsK[ib][row * 64 + sub * 8]),
                 Kb + (size_t)(kb + row) * HD_ + sub * 8);
      async_b128(lds_off(&ldsV[ib][row * 64 + sub * 8]),
                 Vb + (size_t)row * L_ + kb + sub * 8);
    }
  };

  // Resident Q fragments (Q pre-scaled by 1/sqrt(HD) at projection time).
  const __bf16* qrow = Qb + (size_t)(qb + lrow) * HD_;
  const bf16x16 qf0 = frag_a(qrow, 0, lane);
  const bf16x16 qf1 = frag_a(qrow, 32, lane);

  bf16x16 onesf;
#pragma unroll
  for (int i = 0; i < 16; ++i) onesf[i] = (__bf16)1.0f;

  float mrun[8], lrun[8];
#pragma unroll
  for (int r = 0; r < 8; ++r) { mrun[r] = -1e30f; lrun[r] = 0.0f; }
  f32x8 oacc[4] = {};

  issue_tile(0, 0);
  int ib = 0;
  for (int kb = 0; kb < L_; kb += 64, ib ^= 1) {
    wait_async0();        // my chunks of tile(kb) done
    __syncthreads();      // => everyone's chunks done; prev buffer reusable
    if (kb + 64 < L_) issue_tile(ib ^ 1, kb + 64);   // overlap DMA w/ compute

    // ---- S = Qs K^T from LDS K tile ----
    f32x8 st[4];
#pragma unroll
    for (int nt = 0; nt < 4; ++nt) {
      const __bf16* krow = &ldsK[ib][(nt * 16 + lrow) * 64];
      f32x8 s = {};
      s = wmma_bf16(qf0, frag_b(krow, 0, lane), s);
      s = wmma_bf16(qf1, frag_b(krow, 32, lane), s);
      st[nt] = s;
    }

    // ---- online max (16-lane butterfly per half-wave) ----
    float corr[8];
#pragma unroll
    for (int r = 0; r < 8; ++r) {
      float mb = fmaxf(fmaxf(st[0][r], st[1][r]), fmaxf(st[2][r], st[3][r]));
#pragma unroll
      for (int off = 1; off < 16; off <<= 1)
        mb = fmaxf(mb, __shfl_xor(mb, off, 32));
      const float mnew = fmaxf(mrun[r], mb);
      corr[r] = __expf(mrun[r] - mnew);
      mrun[r] = mnew;
    }
#pragma unroll
    for (int nt = 0; nt < 4; ++nt)
#pragma unroll
      for (int r = 0; r < 8; ++r)
        st[nt][r] = __expf(st[nt][r] - mrun[r]);
#pragma unroll
    for (int nt = 0; nt < 4; ++nt)
#pragma unroll
      for (int r = 0; r < 8; ++r) oacc[nt][r] *= corr[r];

    // ---- P (C layout) -> LDS -> A fragments ----
#pragma unroll
    for (int nt = 0; nt < 4; ++nt)
#pragma unroll
      for (int r = 0; r < 8; ++r)
        myp[(half * 8 + r) * 64 + nt * 16 + lrow] = (__bf16)st[nt][r];
    __asm__ volatile("s_wait_dscnt 0" ::: "memory");
    const bf16x16 pf0 = frag_a(myp + (size_t)lrow * 64, 0, lane);
    const bf16x16 pf1 = frag_a(myp + (size_t)lrow * 64, 32, lane);

    // ---- row sums on the matrix pipe: ls = P @ ones ----
    f32x8 ls = {};
    ls = wmma_bf16(pf0, onesf, ls);
    ls = wmma_bf16(pf1, onesf, ls);
#pragma unroll
    for (int r = 0; r < 8; ++r) lrun[r] = lrun[r] * corr[r] + ls[r];

    // ---- O += P * V from LDS V tile ----
#pragma unroll
    for (int nt = 0; nt < 4; ++nt) {
      const __bf16* vrow = &ldsV[ib][(nt * 16 + lrow) * 64];
      oacc[nt] = wmma_bf16(pf0, frag_b(vrow, 0, lane), oacc[nt]);
      oacc[nt] = wmma_bf16(pf1, frag_b(vrow, 32, lane), oacc[nt]);
    }
  }

  // ---- normalize and store ctx[b, l, h*64+hd] (bf16) ----
  const int b = bh >> 4;
  const int h = bh & 15;
#pragma unroll
  for (int nt = 0; nt < 4; ++nt) {
#pragma unroll
    for (int r = 0; r < 8; ++r) {
      const float v = oacc[nt][r] / lrun[r];
      const int l = qb + half * 8 + r;
      ctx[(size_t)(b * L_ + l) * D_ + h * HD_ + nt * 16 + lrow] = (__bf16)v;
    }
  }
}

// --------------------------- launcher --------------------------------------

extern "C" void kernel_launch(void* const* d_in, const int* in_sizes, int n_in,
                              void* d_out, int out_size, void* d_ws, size_t ws_size,
                              hipStream_t stream) {
  const float* x  = (const float*)d_in[0];
  const float* Wq = (const float*)d_in[1];
  const float* Wk = (const float*)d_in[2];
  const float* Wv = (const float*)d_in[3];
  const float* Wo = (const float*)d_in[4];
  const float* bo = (const float*)d_in[5];
  float* out = (float*)d_out;

  char* ws = (char*)d_ws;
  size_t off = 0;
  auto take = [&](size_t bytes) -> void* {
    void* p = ws + off;
    off += (bytes + 255) & ~(size_t)255;
    return p;
  };

  const size_t ntok_elems = (size_t)MTOK * D_;
  const size_t w_elems    = (size_t)D_ * D_;

  __bf16* xb   = (__bf16*)take(ntok_elems * 2);
  __bf16* wqb  = (__bf16*)take(w_elems * 2);
  __bf16* wkb  = (__bf16*)take(w_elems * 2);
  __bf16* wvb  = (__bf16*)take(w_elems * 2);
  __bf16* wob  = (__bf16*)take(w_elems * 2);
  __bf16* qbuf = (__bf16*)take(ntok_elems * 2);   // [b,h,l,hd], pre-scaled
  __bf16* kbuf = (__bf16*)take(ntok_elems * 2);   // [b,h,l,hd]
  __bf16* vtbf = (__bf16*)take(ntok_elems * 2);   // [b,h,hd,l]
  __bf16* ctxb = (__bf16*)take(ntok_elems * 2);   // [b,l,D]

  // 1) convert inputs to bf16
  cvt_f32_bf16_kernel<<<(int)(ntok_elems / 1024), 256, 0, stream>>>(x, xb, (int)ntok_elems);
  cvt_f32_bf16_kernel<<<(int)(w_elems / 1024), 256, 0, stream>>>(Wq, wqb, (int)w_elems);
  cvt_f32_bf16_kernel<<<(int)(w_elems / 1024), 256, 0, stream>>>(Wk, wkb, (int)w_elems);
  cvt_f32_bf16_kernel<<<(int)(w_elems / 1024), 256, 0, stream>>>(Wv, wvb, (int)w_elems);
  cvt_f32_bf16_kernel<<<(int)(w_elems / 1024), 256, 0, stream>>>(Wo, wob, (int)w_elems);

  // 2) projections (fold softmax scale into Q)
  dim3 gg(MTOK / 128, D_ / 128);
  gemm_bf16_kernel<0><<<gg, 128, 0, stream>>>(xb, wqb, qbuf, nullptr, 0.125f);
  gemm_bf16_kernel<0><<<gg, 128, 0, stream>>>(xb, wkb, kbuf, nullptr, 1.0f);
  gemm_bf16_kernel<1><<<gg, 128, 0, stream>>>(xb, wvb, vtbf, nullptr, 1.0f);

  // 3) attention
  attention_kernel<<<(B_ * H_ * (L_ / 16)) / 4, 128, 0, stream>>>(qbuf, kbuf, vtbf, ctxb);

  // 4) output projection + bias -> f32
  gemm_bf16_kernel<2><<<gg, 128, 0, stream>>>(ctxb, wob, out, bo, 1.0f);

  (void)in_sizes; (void)n_in; (void)out_size; (void)ws_size;
}
